// OriginalGraphMixerEncoder_32890859553490
// MI455X (gfx1250) — matrix operand order repaired and checked by reading, not compiled, not run on hardware
//
#include <hip/hip_runtime.h>
#include <hip/hip_bf16.h>

typedef __attribute__((ext_vector_type(16))) _Float16 v16h;
typedef __attribute__((ext_vector_type(8)))  float    v8f;
typedef __attribute__((ext_vector_type(4)))  int      v4i;

#define NSEED 3072
#define TDIM  100
#define TGN   98304

// workspace layout (bytes) — all f16 weights stored TRANSPOSED [n][k]
#define WS_WPT   0u          // 128 x 256
#define WS_T1T   65536u      // 2 x 32 x 64
#define WS_T2T   73728u      // 2 x 64 x 32
#define WS_C1T   81920u      // 2 x 512 x 128
#define WS_C2T   344064u     // 2 x 128 x 512
#define WS_OWT   606208u     // 128 x 256
#define WS_ZBAR  671744u     // 3072*128 f32
#define WS_TGA   2244608u    // 3072*128 f32
#define WS_CNT   3817472u    // 3072 f32

typedef __attribute__((address_space(1))) v4i gv4i;
typedef __attribute__((address_space(3))) v4i lv4i;

#define ASYNC_WAIT0() asm volatile("s_wait_asynccnt 0x0" ::: "memory")
#define ASYNC_WAIT8() asm volatile("s_wait_asynccnt 0x8" ::: "memory")

__device__ __forceinline__ void async_cp16(_Float16* lds_dst, const _Float16* gsrc) {
  __builtin_amdgcn_global_load_async_to_lds_b128((gv4i*)gsrc, (lv4i*)lds_dst, 0, 0);
}

// copy rows of a tile: contiguous rowHalfs halves per row, 16B chunks, async to LDS
__device__ __forceinline__ void stage_async(_Float16* dst, int dstStride,
                                            const _Float16* src, int srcStride,
                                            int rows, int rowHalfs, int tid) {
  int cpr = rowHalfs >> 3;            // 16-byte chunks per row
  int total = rows * cpr;
  for (int i = tid; i < total; i += 256) {
    int r = i / cpr, c = (i % cpr) << 3;
    async_cp16(dst + r * dstStride + c, src + r * srcStride + c);
  }
}

__device__ __forceinline__ v8f zero8() {
  v8f z;
#pragma unroll
  for (int i = 0; i < 8; ++i) z[i] = 0.0f;
  return z;
}

__device__ __forceinline__ v8f wmma16(v16h a, v16h b, v8f c) {
  return __builtin_amdgcn_wmma_f32_16x16x32_f16(false, a, false, b, (short)0, c, false, false);
}

// A fragment (16x32 f16) from row-major LDS [m][k]
__device__ __forceinline__ v16h ldA(const _Float16* s, int ld, int m0, int k0, int lane) {
  v16h a;
  const _Float16* p = s + (m0 + (lane & 15)) * ld + k0 + ((lane >> 4) << 4);
#pragma unroll
  for (int i = 0; i < 8; ++i) { a[i] = p[i]; a[i + 8] = p[i + 8]; }
  return a;
}

// B fragment (32x16 f16) from row-major LDS [k][n]
__device__ __forceinline__ v16h ldB(const _Float16* s, int ld, int k0, int n0, int lane) {
  v16h b;
  const _Float16* p = s + (k0 + ((lane >> 4) << 4)) * ld + n0 + (lane & 15);
#pragma unroll
  for (int h = 0; h < 16; ++h) b[h] = p[h * ld];
  return b;
}

// B fragment from TRANSPOSED LDS [n][k] (contiguous per lane)
__device__ __forceinline__ v16h ldBt(const _Float16* s, int ld, int k0, int n0, int lane) {
  v16h b;
  const _Float16* p = s + (n0 + (lane & 15)) * ld + k0 + ((lane >> 4) << 4);
#pragma unroll
  for (int h = 0; h < 16; ++h) b[h] = p[h];
  return b;
}

__device__ __forceinline__ float gelu_t(float x) {
  float x3 = x * x * x;
  return 0.5f * x * (1.0f + tanhf(0.7978845608f * (x + 0.044715f * x3)));
}

// ---------------- weight conversion (f32 -> f16, transposed) ----------------
__global__ void cvt_weights(const float* projW, const float* tw1, const float* tw2,
                            const float* cw1, const float* cw2, const float* outw,
                            _Float16* wpT, _Float16* t1T, _Float16* t2T,
                            _Float16* c1T, _Float16* c2T, _Float16* owT) {
  int tid = blockIdx.x * blockDim.x + threadIdx.x;
  int nt = gridDim.x * blockDim.x;
  for (int i = tid; i < 128 * 256; i += nt) {          // wpT[c][k]
    int c = i >> 8, k = i & 255;
    wpT[i] = (k < 228) ? (_Float16)projW[k * 128 + c] : (_Float16)0.0f;
  }
  for (int i = tid; i < 2 * 32 * 64; i += nt) {        // t1T[l][h][k]
    int l = i >> 11, r = i & 2047, h = r >> 6, k = r & 63;
    t1T[i] = (_Float16)tw1[l * 2048 + k * 32 + h];
  }
  for (int i = tid; i < 2 * 64 * 32; i += nt) {        // t2T[l][k][h]
    int l = i >> 11, r = i & 2047, k = r >> 5, h = r & 31;
    t2T[i] = (_Float16)tw2[l * 2048 + h * 64 + k];
  }
  for (int i = tid; i < 2 * 512 * 128; i += nt) {      // c1T[l][n][k]
    int l = i >> 16, r = i & 65535, nn = r >> 7, k = r & 127;
    c1T[i] = (_Float16)cw1[l * 65536 + k * 512 + nn];
  }
  for (int i = tid; i < 2 * 128 * 512; i += nt) {      // c2T[l][n][k]
    int l = i >> 16, r = i & 65535, nn = r >> 9, k = r & 511;
    c2T[i] = (_Float16)cw2[l * 65536 + k * 128 + nn];
  }
  for (int i = tid; i < 128 * 256; i += nt) {          // owT[c][k]
    int c = i >> 8, k = i & 255;
    owT[i] = (_Float16)outw[k * 128 + c];
  }
}

__global__ void zero_f32(float* p, int n) {
  int i = blockIdx.x * blockDim.x + threadIdx.x;
  if (i < n) p[i] = 0.0f;
}

// ---------------- ragged segment accumulation ----------------
__global__ void seg_accum(const int* tg_idx, const int* tg_seg, const float* node_feat,
                          float* tg_acc, float* cnt) {
  int gid = blockIdx.x * blockDim.x + threadIdx.x;
  int i = gid >> 7;
  int c = gid & 127;
  if (i < TGN) {
    int seg = tg_seg[i];
    int idx = tg_idx[i];
    atomicAdd(&tg_acc[(size_t)seg * 128 + c], node_feat[(size_t)idx * 128 + c]);
    if (c == 0) atomicAdd(&cnt[seg], 1.0f);
  }
}

// ---------------- fused mixer ----------------
__device__ __forceinline__ void layernorm_block(const float* sZ, _Float16* sX,
                                                float* sMean, float* sRstd,
                                                const float* g, const float* b, int tid) {
  if (tid < 64) {
    float s = 0.0f, s2 = 0.0f;
    for (int c = 0; c < 128; ++c) {
      float v = sZ[tid * 132 + c];
      s += v; s2 += v * v;
    }
    float m = s * (1.0f / 128.0f);
    float var = s2 * (1.0f / 128.0f) - m * m;
    sMean[tid] = m;
    sRstd[tid] = rsqrtf(var + 1e-5f);
  }
  __syncthreads();
  for (int i = tid; i < 64 * 128; i += 256) {
    int r = i >> 7, c = i & 127;
    float v = (sZ[r * 132 + c] - sMean[r]) * sRstd[r] * g[c] + b[c];
    sX[r * 136 + c] = (_Float16)v;
  }
  __syncthreads();
}

__global__ __launch_bounds__(256)
void mixer_kernel(const float* edge_feat, const float* seed_times, const float* nbr_time,
                  const float* time_w, const float* proj_b,
                  const float* ln1_g, const float* ln1_b,
                  const float* tb1, const float* tb2,
                  const float* ln2_g, const float* ln2_b,
                  const float* cb1, const float* cb2,
                  const int* nbr_nids,
                  const _Float16* wpT, const _Float16* t1T, const _Float16* t2T,
                  const _Float16* c1T, const _Float16* c2T,
                  float* zbar) {
  __shared__ __align__(16) float    sZ[64 * 132];       // f32 master activation
  __shared__ __align__(16) _Float16 sX[64 * 136];       // f16 LN'ed activation
  __shared__ __align__(16) _Float16 sWbuf[2][128 * 136];// ping-pong weight tiles
  __shared__ __align__(16) _Float16 sP[64 * 264];       // panel / channel hidden block
  __shared__ float sMean[64], sRstd[64];

  int tid = threadIdx.x;
  int lane = tid & 31;
  int wv = tid >> 5;
  int n = blockIdx.x;

  // ---- prefetch both projection weight halves (async, transposed rows)
  stage_async(sWbuf[0], 136, wpT + 0,   256, 128, 128, tid);
  stage_async(sWbuf[1], 136, wpT + 128, 256, 128, 128, tid);

  // ---- meanwhile build [64, 228->256] f16 panel with VALU
  const float* ef = edge_feat + (size_t)n * 64 * 128;
  for (int i = tid; i < 64 * 128; i += 256) {
    int r = i >> 7, c = i & 127;
    sP[r * 264 + c] = (_Float16)ef[i];
  }
  float st = seed_times[n];
  for (int i = tid; i < 64 * 128; i += 256) {
    int r = i >> 7, c = i & 127;
    float v = 0.0f;
    if (c < TDIM) v = cosf((st - nbr_time[n * 64 + r]) * time_w[c]);
    sP[r * 264 + 128 + c] = (_Float16)v;
  }

  // ---- projection GEMM: Z[64,128] = P[64,256] @ Wp[256,128]
  v8f acc[4];
#pragma unroll
  for (int t = 0; t < 4; ++t) acc[t] = zero8();
  ASYNC_WAIT8();         // half 0 landed (half 1 may be in flight)
  __syncthreads();
#pragma unroll
  for (int kh = 0; kh < 2; ++kh) {
#pragma unroll
    for (int t = 0; t < 4; ++t) {
      int tile = wv + t * 8, mt = tile & 3, nt = tile >> 2;
#pragma unroll
      for (int kk = 0; kk < 4; ++kk)
        acc[t] = wmma16(ldA(sP, 264, mt * 16, kh * 128 + kk * 32, lane),
                        ldBt(sWbuf[kh], 136, kk * 32, nt * 16, lane), acc[t]);
    }
    if (kh == 0) { ASYNC_WAIT0(); __syncthreads(); }
  }
#pragma unroll
  for (int t = 0; t < 4; ++t) {
    int tile = wv + t * 8, mt = tile & 3, nt = tile >> 2;
    int c = nt * 16 + (lane & 15);
    float bias = proj_b[c];
#pragma unroll
    for (int r = 0; r < 8; ++r) {
      int m = mt * 16 + ((lane >> 4) << 3) + r;
      sZ[m * 132 + c] = acc[t][r] + bias;
    }
  }
  __syncthreads();

  // token scratch lives inside sWbuf[0]
  _Float16* sT1 = sWbuf[0];          // [32][72]
  _Float16* sT2 = sWbuf[0] + 2304;   // [64][40]
  _Float16* sH1 = sWbuf[0] + 4864;   // [32][136]
  _Float16* sHb = sP;                // [64][136] channel hidden block

  for (int l = 0; l < 2; ++l) {
    // prefetch token weights (buf0) + channel W1_0 (buf1) while LN runs
    stage_async(sT1, 72, t1T + l * 2048, 64, 32, 64, tid);
    stage_async(sT2, 40, t2T + l * 2048, 32, 64, 32, tid);
    stage_async(sWbuf[1], 136, c1T + (size_t)l * 65536, 128, 128, 128, tid);

    layernorm_block(sZ, sX, sMean, sRstd, ln1_g + l * 128, ln1_b + l * 128, tid);
    ASYNC_WAIT8();     // token weights done (W1_0 still in flight)
    __syncthreads();

    // token GEMM1: H1[32,128] = tW1^T[32,64] @ X[64,128], +tb1, gelu
#pragma unroll
    for (int t = 0; t < 2; ++t) {
      int tile = wv + t * 8, mt = tile & 1, nt = tile >> 1;
      v8f d = zero8();
#pragma unroll
      for (int kk = 0; kk < 2; ++kk)
        d = wmma16(ldA(sT1, 72, mt * 16, kk * 32, lane),
                   ldB(sX, 136, kk * 32, nt * 16, lane), d);
      int nb = nt * 16 + (lane & 15);
#pragma unroll
      for (int r = 0; r < 8; ++r) {
        int m = mt * 16 + ((lane >> 4) << 3) + r;
        sH1[m * 136 + nb] = (_Float16)gelu_t(d[r] + tb1[l * 32 + m]);
      }
    }
    __syncthreads();

    // token GEMM2: Z += tW2^T[64,32] @ H1[32,128] + tb2
#pragma unroll
    for (int t = 0; t < 4; ++t) {
      int tile = wv + t * 8, mt = tile & 3, nt = tile >> 2;
      v8f d = wmma16(ldA(sT2, 40, mt * 16, 0, lane),
                     ldB(sH1, 136, 0, nt * 16, lane), zero8());
      int nb = nt * 16 + (lane & 15);
#pragma unroll
      for (int r = 0; r < 8; ++r) {
        int m = mt * 16 + ((lane >> 4) << 3) + r;
        sZ[m * 132 + nb] += d[r] + tb2[l * 64 + m];
      }
    }
    __syncthreads();

    layernorm_block(sZ, sX, sMean, sRstd, ln2_g + l * 128, ln2_b + l * 128, tid);

    // channel MLP 128->512->128: 8-step pipeline (W1_0,W2_0,...,W2_3), ping-pong bufs
    v8f acc2[4];
#pragma unroll
    for (int t = 0; t < 4; ++t) acc2[t] = zero8();
    for (int s = 0; s < 8; ++s) {
      _Float16* cur = sWbuf[(s + 1) & 1];
      if (s < 7) {
        int jn = (s + 1) >> 1;
        _Float16* nxt = sWbuf[s & 1];
        if ((s + 1) & 1)   // odd step -> W2_jn : c2T[l][n=128][k slice jn*128..]
          stage_async(nxt, 136, c2T + (size_t)l * 65536 + jn * 128, 512, 128, 128, tid);
        else               // even step -> W1_jn : c1T[l][n slice jn*128..][k=128]
          stage_async(nxt, 136, c1T + (size_t)l * 65536 + (size_t)jn * 128 * 128, 128, 128, 128, tid);
        ASYNC_WAIT8();
      } else {
        ASYNC_WAIT0();
      }
      __syncthreads();
      int j = s >> 1;
      if ((s & 1) == 0) {
        // GEMM1: hid_j[64,128] = X @ W1_j, +cb1, gelu -> sHb
#pragma unroll
        for (int t = 0; t < 4; ++t) {
          int tile = wv + t * 8, mt = tile & 3, nt = tile >> 2;
          v8f d = zero8();
#pragma unroll
          for (int kk = 0; kk < 4; ++kk)
            d = wmma16(ldA(sX, 136, mt * 16, kk * 32, lane),
                       ldBt(cur, 136, kk * 32, nt * 16, lane), d);
          int nb = nt * 16 + (lane & 15);
          float b1 = cb1[l * 512 + j * 128 + nb];
#pragma unroll
          for (int r = 0; r < 8; ++r) {
            int m = mt * 16 + ((lane >> 4) << 3) + r;
            sHb[m * 136 + nb] = (_Float16)gelu_t(d[r] + b1);
          }
        }
      } else {
        // GEMM2: acc2 += sHb @ W2_j
#pragma unroll
        for (int t = 0; t < 4; ++t) {
          int tile = wv + t * 8, mt = tile & 3, nt = tile >> 2;
#pragma unroll
          for (int kk = 0; kk < 4; ++kk)
            acc2[t] = wmma16(ldA(sHb, 136, mt * 16, kk * 32, lane),
                             ldBt(cur, 136, kk * 32, nt * 16, lane), acc2[t]);
        }
      }
      __syncthreads();
    }
#pragma unroll
    for (int t = 0; t < 4; ++t) {
      int tile = wv + t * 8, mt = tile & 3, nt = tile >> 2;
      int nb = nt * 16 + (lane & 15);
      float b2 = cb2[l * 128 + nb];
#pragma unroll
      for (int r = 0; r < 8; ++r) {
        int m = mt * 16 + ((lane >> 4) << 3) + r;
        sZ[m * 132 + nb] += acc2[t][r] + b2;
      }
    }
    __syncthreads();
  }

  // ---- masked mean over tokens -> zbar[n, 128]
  if (tid < 128) {
    float s = 0.0f, ct = 0.0f;
    for (int r = 0; r < 64; ++r) {
      float mk = (nbr_nids[n * 64 + r] != 0) ? 1.0f : 0.0f;
      s += sZ[r * 132 + tid] * mk;
      ct += mk;
    }
    zbar[(size_t)n * 128 + tid] = s / fmaxf(ct, 1.0f);
  }
}

// ---------------- final output GEMM: [64,256] @ outW[256,128] ----------------
__global__ __launch_bounds__(256)
void out_gemm(const float* zbar, const float* tg_acc, const float* cnt,
              const float* node_feat, const int* seed_ids, const _Float16* owT,
              const float* out_b, float* out) {
  __shared__ __align__(16) _Float16 fA[64 * 264];
  __shared__ __align__(16) _Float16 fW[2][128 * 136];
  int tid = threadIdx.x, lane = tid & 31, wv = tid >> 5;
  int row0 = blockIdx.x * 64;

  stage_async(fW[0], 136, owT + 0,   256, 128, 128, tid);
  stage_async(fW[1], 136, owT + 128, 256, 128, 128, tid);

  for (int i = tid; i < 64 * 128; i += 256) {
    int r = i >> 7, c = i & 127;
    fA[r * 264 + c] = (_Float16)zbar[(size_t)(row0 + r) * 128 + c];
  }
  for (int i = tid; i < 64 * 128; i += 256) {
    int r = i >> 7, c = i & 127;
    int n = row0 + r;
    float ct = fmaxf(cnt[n], 1.0f);
    float v = tg_acc[(size_t)n * 128 + c] / ct +
              node_feat[(size_t)seed_ids[n] * 128 + c];
    fA[r * 264 + 128 + c] = (_Float16)v;
  }

  v8f acc[4];
#pragma unroll
  for (int t = 0; t < 4; ++t) acc[t] = zero8();
  ASYNC_WAIT8();
  __syncthreads();
#pragma unroll
  for (int kh = 0; kh < 2; ++kh) {
#pragma unroll
    for (int t = 0; t < 4; ++t) {
      int tile = wv + t * 8, mt = tile & 3, nt = tile >> 2;
#pragma unroll
      for (int kk = 0; kk < 4; ++kk)
        acc[t] = wmma16(ldA(fA, 264, mt * 16, kh * 128 + kk * 32, lane),
                        ldBt(fW[kh], 136, kk * 32, nt * 16, lane), acc[t]);
    }
    if (kh == 0) { ASYNC_WAIT0(); __syncthreads(); }
  }
#pragma unroll
  for (int t = 0; t < 4; ++t) {
    int tile = wv + t * 8, mt = tile & 3, nt = tile >> 2;
    int c = nt * 16 + (lane & 15);
    float b = out_b[c];
#pragma unroll
    for (int r = 0; r < 8; ++r) {
      int m = mt * 16 + ((lane >> 4) << 3) + r;
      out[(size_t)(row0 + m) * 128 + c] = acc[t][r] + b;
    }
  }
}

extern "C" void kernel_launch(void* const* d_in, const int* in_sizes, int n_in,
                              void* d_out, int out_size, void* d_ws, size_t ws_size,
                              hipStream_t stream) {
  (void)in_sizes; (void)n_in; (void)out_size; (void)ws_size;
  const float* edge_feat  = (const float*)d_in[0];
  const float* seed_times = (const float*)d_in[1];
  const float* nbr_time   = (const float*)d_in[2];
  const float* node_feat  = (const float*)d_in[3];
  const float* time_w     = (const float*)d_in[4];
  const float* proj_W     = (const float*)d_in[5];
  const float* proj_b     = (const float*)d_in[6];
  const float* ln1_g      = (const float*)d_in[7];
  const float* ln1_b      = (const float*)d_in[8];
  const float* tW1        = (const float*)d_in[9];
  const float* tb1        = (const float*)d_in[10];
  const float* tW2        = (const float*)d_in[11];
  const float* tb2        = (const float*)d_in[12];
  const float* ln2_g      = (const float*)d_in[13];
  const float* ln2_b      = (const float*)d_in[14];
  const float* cW1        = (const float*)d_in[15];
  const float* cb1        = (const float*)d_in[16];
  const float* cW2        = (const float*)d_in[17];
  const float* cb2        = (const float*)d_in[18];
  const float* out_W      = (const float*)d_in[19];
  const float* out_b      = (const float*)d_in[20];
  const int* nbr_nids     = (const int*)d_in[21];
  const int* tg_idx       = (const int*)d_in[22];
  const int* tg_seg       = (const int*)d_in[23];
  const int* seed_ids     = (const int*)d_in[24];

  char* ws = (char*)d_ws;
  _Float16* wpT  = (_Float16*)(ws + WS_WPT);
  _Float16* t1T  = (_Float16*)(ws + WS_T1T);
  _Float16* t2T  = (_Float16*)(ws + WS_T2T);
  _Float16* c1T  = (_Float16*)(ws + WS_C1T);
  _Float16* c2T  = (_Float16*)(ws + WS_C2T);
  _Float16* owT  = (_Float16*)(ws + WS_OWT);
  float* zbar    = (float*)(ws + WS_ZBAR);
  float* tga     = (float*)(ws + WS_TGA);
  float* cnt     = (float*)(ws + WS_CNT);

  cvt_weights<<<512, 256, 0, stream>>>(proj_W, tW1, tW2, cW1, cW2, out_W,
                                       wpT, t1T, t2T, c1T, c2T, owT);
  int nz = NSEED * 128 + NSEED;   // tg_acc and cnt contiguous
  zero_f32<<<(nz + 255) / 256, 256, 0, stream>>>(tga, nz);
  mixer_kernel<<<NSEED, 256, 0, stream>>>(edge_feat, seed_times, nbr_time, time_w, proj_b,
                                          ln1_g, ln1_b, tb1, tb2, ln2_g, ln2_b, cb1, cb2,
                                          nbr_nids, wpT, t1T, t2T, c1T, c2T, zbar);
  seg_accum<<<(TGN * 128) / 256, 256, 0, stream>>>(tg_idx, tg_seg, node_feat, tga, cnt);
  out_gemm<<<NSEED / 64, 256, 0, stream>>>(zbar, tga, cnt, node_feat, seed_ids,
                                           owT, out_b, (float*)d_out);
}
